// CausalSelfAttention_26568667693347
// MI455X (gfx1250) — matrix operand order repaired and checked by reading, not compiled, not run on hardware
//
#include <hip/hip_runtime.h>

// ---------------------------------------------------------------------------
// CausalSelfAttention  B=8, T=1024, C=768, NH=12, HS=64  (gfx1250 / CDNA5)
// bf16 WMMA (f32 accumulate) for all three matmuls + transposed flash-attn.
// Round 2: 64x64 GEMM wave tiles (16 WMMA / 8 frag loads), 32-query flash
// wave tiles (16 WMMA / 8 frag loads per 32-key step).
// ---------------------------------------------------------------------------

typedef __attribute__((ext_vector_type(16))) __bf16        bf16x16;
typedef __attribute__((ext_vector_type(8)))  float         f32x8;
typedef __attribute__((ext_vector_type(4)))  unsigned int  u32x4;

union Frag {
    u32x4          u4[2];
    unsigned short s[16];
    bf16x16        v;
};

__device__ __forceinline__ unsigned short f2bf(float f) {
    unsigned int u = __float_as_uint(f);
    unsigned int r = u + 0x7FFFu + ((u >> 16) & 1u);   // round-to-nearest-even
    return (unsigned short)(r >> 16);
}

__device__ __forceinline__ f32x8 wmma_bf16(bf16x16 a, bf16x16 b, f32x8 c) {
    // (neg_a, A, neg_b, B, c_mod, C, reuse_a, reuse_b)
    return __builtin_amdgcn_wmma_f32_16x16x32_bf16(false, a, false, b, (short)0, c, false, false);
}

// A-matrix fragment (16x32, row-major source, leading dim ld):
// lane<16: K = k0..k0+15 ; lane>=16: K = k0+16..k0+31
__device__ __forceinline__ void load_fragA(Frag& f, const unsigned short* base,
                                           int row0, int k0, int ld, int lane) {
    int r = lane & 15, hi = lane >> 4;
    const unsigned short* p = base + (size_t)(row0 + r) * ld + (k0 + hi * 16);
    f.u4[0] = *(const u32x4*)p;
    f.u4[1] = *(const u32x4*)(p + 8);
}

// B-matrix fragment (32x16).  Source stored column-major-as-rows: [N, K].
// lane<16: K = k0..k0+15 ; lane>=16: K = k0+16..k0+31  (contiguous 32 B)
__device__ __forceinline__ void load_fragB(Frag& f, const unsigned short* base,
                                           int col0, int k0, int ld, int lane) {
    int r = lane & 15, hi = lane >> 4;
    const unsigned short* p = base + (size_t)(col0 + r) * ld + (k0 + hi * 16);
    f.u4[0] = *(const u32x4*)p;
    f.u4[1] = *(const u32x4*)(p + 8);
}

// ------------------------------- converts ----------------------------------

__global__ void cvt_f32_bf16(const float* __restrict__ in, unsigned short* __restrict__ out, int n) {
    int i = blockIdx.x * blockDim.x + threadIdx.x;
    if (i < n) out[i] = f2bf(in[i]);
}

// in: [K, N] f32 row-major  ->  out: [N, K] bf16 row-major
__global__ void transpose_w(const float* __restrict__ in, unsigned short* __restrict__ out,
                            int K, int N) {
    int i = blockIdx.x * blockDim.x + threadIdx.x;
    if (i < K * N) {
        int nn = i / K, kk = i - nn * K;
        out[i] = f2bf(in[(size_t)kk * N + nn]);
    }
}

// ------------------------- kernel 1: QKV GEMM ------------------------------
// [8192 x 768] x [768 x 2304] (wT stored [2304 x 768]).  Wave tile 64x64.
// Epilogue: +bias, Q*=0.125, scatter Q,K -> [B,NH,T,HS] bf16, V -> [B,NH,HS,T].

__global__ __launch_bounds__(256) void qkv_gemm_kernel(
    const unsigned short* __restrict__ xb, const unsigned short* __restrict__ wT,
    const float* __restrict__ bias,
    unsigned short* __restrict__ qb, unsigned short* __restrict__ kb,
    unsigned short* __restrict__ vb) {
    const int lane = threadIdx.x & 31;
    const int w    = threadIdx.x >> 5;
    const int m0   = blockIdx.y * 256 + (w & 3) * 64;
    const int n0   = blockIdx.x * 128 + (w >> 2) * 64;

    const f32x8 z = {0.f,0.f,0.f,0.f,0.f,0.f,0.f,0.f};
    f32x8 acc[4][4];
#pragma unroll
    for (int i = 0; i < 4; ++i)
#pragma unroll
        for (int j = 0; j < 4; ++j) acc[i][j] = z;

    for (int k0 = 0; k0 < 768; k0 += 32) {
        Frag aF[4], bF[4];
#pragma unroll
        for (int i = 0; i < 4; ++i) load_fragA(aF[i], xb, m0 + i * 16, k0, 768, lane);
#pragma unroll
        for (int j = 0; j < 4; ++j) load_fragB(bF[j], wT, n0 + j * 16, k0, 768, lane);
#pragma unroll
        for (int i = 0; i < 4; ++i)
#pragma unroll
            for (int j = 0; j < 4; ++j)
                acc[i][j] = wmma_bf16(aF[i].v, bF[j].v, acc[i][j]);
    }

    const int r = lane & 15, hi = lane >> 4;
#pragma unroll
    for (int j = 0; j < 4; ++j) {
        const int n    = n0 + j * 16 + r;
        const float bv = bias[n];
        const int seg  = n / 768;          // 0=Q 1=K 2=V
        const int c    = n - seg * 768;
        const int h    = c >> 6, d = c & 63;
#pragma unroll
        for (int i = 0; i < 4; ++i) {
#pragma unroll
            for (int vr = 0; vr < 8; ++vr) {
                const int m = m0 + i * 16 + vr + hi * 8;
                const int b = m >> 10, t = m & 1023;
                const float val = acc[i][j][vr] + bv;
                const size_t hd = (size_t)(b * 12 + h);
                if (seg == 0)      qb[(hd * 1024 + t) * 64 + d] = f2bf(val * 0.125f);
                else if (seg == 1) kb[(hd * 1024 + t) * 64 + d] = f2bf(val);
                else               vb[(hd * 64 + d) * 1024 + t] = f2bf(val);
            }
        }
    }
}

// ---------------------- kernel 2: flash attention --------------------------
// 32 queries (two 16-query sub-tiles) per wave.  Computes S^T = K * Q^T so
// each lane owns one query column; online softmax = in-lane reduce + xor(16).
// Y^T = V^T * P^T; V stored [HS, T]; K/V fragments reused by both q-subtiles.

__global__ __launch_bounds__(256) void flash_kernel(
    const unsigned short* __restrict__ qb, const unsigned short* __restrict__ kb,
    const unsigned short* __restrict__ vb, unsigned short* __restrict__ yb) {
    const int lane = threadIdx.x & 31;
    const int w    = threadIdx.x >> 5;
    const int g    = blockIdx.x * 8 + w;      // 0..3071
    const int bh   = g >> 5;                  // B*NH = 96
    const int q0   = (g & 31) << 5;           // 32 queries per wave
    const int r = lane & 15, hi = lane >> 4;

    const unsigned short* qp = qb + (size_t)bh * 1024 * 64;
    const unsigned short* kp = kb + (size_t)bh * 1024 * 64;
    const unsigned short* vp = vb + (size_t)bh * 64 * 1024;

    // Q^T B-fragments [qsub][d-chunk], resident across the K loop.
    Frag bQ[2][2];
#pragma unroll
    for (int qs = 0; qs < 2; ++qs) {
#pragma unroll
        for (int c = 0; c < 2; ++c) {
            const unsigned short* p = qp + (size_t)(q0 + qs * 16 + r) * 64 + c * 32 + hi * 16;
            bQ[qs][c].u4[0] = *(const u32x4*)p;
            bQ[qs][c].u4[1] = *(const u32x4*)(p + 8);
        }
    }

    const f32x8 z = {0.f,0.f,0.f,0.f,0.f,0.f,0.f,0.f};
    f32x8 yacc[4][2];                          // [d-tile][qsub]
#pragma unroll
    for (int dt = 0; dt < 4; ++dt) { yacc[dt][0] = z; yacc[dt][1] = z; }

    float m_run[2] = {-1e30f, -1e30f};
    float l_run[2] = {0.f, 0.f};

    for (int k0 = 0; k0 < q0 + 32; k0 += 32) {
        // S^T tiles: [key-subtile][qsub]; K-fragments reused for both qsubs.
        f32x8 st[2][2];
#pragma unroll
        for (int kt = 0; kt < 2; ++kt) {
            Frag aK[2];
#pragma unroll
            for (int c = 0; c < 2; ++c) {
                const unsigned short* p = kp + (size_t)(k0 + kt * 16 + r) * 64 + c * 32 + hi * 16;
                aK[c].u4[0] = *(const u32x4*)p;
                aK[c].u4[1] = *(const u32x4*)(p + 8);
            }
#pragma unroll
            for (int qs = 0; qs < 2; ++qs) {
                f32x8 s = z;
                s = wmma_bf16(aK[0].v, bQ[qs][0].v, s);
                s = wmma_bf16(aK[1].v, bQ[qs][1].v, s);
                st[kt][qs] = s;
            }
        }

        // online softmax per q-subtile
        float scale[2];
        Frag bP[2];
#pragma unroll
        for (int qs = 0; qs < 2; ++qs) {
            const int q_lane = q0 + qs * 16 + r;
            float mloc = -1e30f;
#pragma unroll
            for (int kt = 0; kt < 2; ++kt) {
#pragma unroll
                for (int vr = 0; vr < 8; ++vr) {
                    const int key = k0 + kt * 16 + vr + hi * 8;
                    const float sv = (key <= q_lane) ? st[kt][qs][vr] : -1e30f;
                    st[kt][qs][vr] = sv;
                    mloc = fmaxf(mloc, sv);
                }
            }
            mloc = fmaxf(mloc, __shfl_xor(mloc, 16, 32));
            const float m_new = fmaxf(m_run[qs], mloc);
            scale[qs] = __expf(m_run[qs] - m_new);

            float psum = 0.f;
#pragma unroll
            for (int kt = 0; kt < 2; ++kt) {
#pragma unroll
                for (int vr = 0; vr < 8; ++vr) {
                    const float p = __expf(st[kt][qs][vr] - m_new);
                    st[kt][qs][vr] = p;
                    psum += p;
                }
            }
            psum += __shfl_xor(psum, 16, 32);
            l_run[qs] = l_run[qs] * scale[qs] + psum;
            m_run[qs] = m_new;

            // Pack P^T B-fragment: lane<16 keys 0..15, lane>=16 keys 16..31.
#pragma unroll
            for (int vr = 0; vr < 8; ++vr) {
                const float x0 = __shfl_xor(st[0][qs][vr], 16, 32);  // partner tile0
                const float x1 = __shfl_xor(st[1][qs][vr], 16, 32);  // partner tile1
                bP[qs].s[vr]     = f2bf(hi ? x1 : st[0][qs][vr]);
                bP[qs].s[8 + vr] = f2bf(hi ? st[1][qs][vr] : x0);
            }
        }

        // Y^T += V^T * P^T  (4 d-tiles; each V-fragment reused by both qsubs)
#pragma unroll
        for (int dt = 0; dt < 4; ++dt) {
            Frag aV;
            const unsigned short* p = vp + (size_t)(dt * 16 + r) * 1024 + k0 + hi * 16;
            aV.u4[0] = *(const u32x4*)p;
            aV.u4[1] = *(const u32x4*)(p + 8);
#pragma unroll
            for (int qs = 0; qs < 2; ++qs) {
#pragma unroll
                for (int vr = 0; vr < 8; ++vr) yacc[dt][qs][vr] *= scale[qs];
                yacc[dt][qs] = wmma_bf16(aV.v, bP[qs].v, yacc[dt][qs]);
            }
        }
    }

    const int b = bh / 12, h = bh - (bh / 12) * 12;
#pragma unroll
    for (int qs = 0; qs < 2; ++qs) {
        const float inv = 1.0f / l_run[qs];
        const size_t rowbase = ((size_t)b * 1024 + q0 + qs * 16 + r) * 768 + h * 64;
#pragma unroll
        for (int dt = 0; dt < 4; ++dt) {
#pragma unroll
            for (int vr = 0; vr < 8; ++vr) {
                const int d = dt * 16 + vr + hi * 8;
                yb[rowbase + d] = f2bf(yacc[dt][qs][vr] * inv);
            }
        }
    }
}

// ---------------------- kernel 3: output projection ------------------------
// [8192 x 768] x [768 x 768] (wT stored [768 x 768]) + bias -> f32 d_out.

__global__ __launch_bounds__(256) void proj_gemm_kernel(
    const unsigned short* __restrict__ yb, const unsigned short* __restrict__ wT,
    const float* __restrict__ bias, float* __restrict__ out) {
    const int lane = threadIdx.x & 31;
    const int w    = threadIdx.x >> 5;
    const int m0   = blockIdx.y * 256 + (w & 3) * 64;
    const int n0   = blockIdx.x * 128 + (w >> 2) * 64;

    const f32x8 z = {0.f,0.f,0.f,0.f,0.f,0.f,0.f,0.f};
    f32x8 acc[4][4];
#pragma unroll
    for (int i = 0; i < 4; ++i)
#pragma unroll
        for (int j = 0; j < 4; ++j) acc[i][j] = z;

    for (int k0 = 0; k0 < 768; k0 += 32) {
        Frag aF[4], bF[4];
#pragma unroll
        for (int i = 0; i < 4; ++i) load_fragA(aF[i], yb, m0 + i * 16, k0, 768, lane);
#pragma unroll
        for (int j = 0; j < 4; ++j) load_fragB(bF[j], wT, n0 + j * 16, k0, 768, lane);
#pragma unroll
        for (int i = 0; i < 4; ++i)
#pragma unroll
            for (int j = 0; j < 4; ++j)
                acc[i][j] = wmma_bf16(aF[i].v, bF[j].v, acc[i][j]);
    }

    const int r = lane & 15, hi = lane >> 4;
#pragma unroll
    for (int j = 0; j < 4; ++j) {
        const int n    = n0 + j * 16 + r;
        const float bv = bias[n];
#pragma unroll
        for (int i = 0; i < 4; ++i) {
#pragma unroll
            for (int vr = 0; vr < 8; ++vr) {
                const int m = m0 + i * 16 + vr + hi * 8;
                out[(size_t)m * 768 + n] = acc[i][j][vr] + bv;
            }
        }
    }
}

// ------------------------------- launcher ----------------------------------

extern "C" void kernel_launch(void* const* d_in, const int* in_sizes, int n_in,
                              void* d_out, int out_size, void* d_ws, size_t ws_size,
                              hipStream_t stream) {
    const float* x      = (const float*)d_in[0];   // [8,1024,768]
    const float* w_attn = (const float*)d_in[1];   // [768,2304]
    const float* b_attn = (const float*)d_in[2];   // [2304]
    const float* w_proj = (const float*)d_in[3];   // [768,768]
    const float* b_proj = (const float*)d_in[4];   // [768]
    float* out = (float*)d_out;

    constexpr size_t SZ_XB     = (size_t)8192 * 768 * 2;   // x bf16
    constexpr size_t SZ_WQKVT  = (size_t)2304 * 768 * 2;   // w_attn^T bf16
    constexpr size_t SZ_WPROJT = (size_t)768  * 768 * 2;   // w_proj^T bf16
    constexpr size_t SZ_HEADS  = (size_t)8192 * 768 * 2;   // q / k / v / y each

    char* p = (char*)d_ws;
    unsigned short* xb     = (unsigned short*)p; p += SZ_XB;
    unsigned short* wqkvT  = (unsigned short*)p; p += SZ_WQKVT;
    unsigned short* wprojT = (unsigned short*)p; p += SZ_WPROJT;
    unsigned short* qb     = (unsigned short*)p; p += SZ_HEADS;
    unsigned short* kb     = (unsigned short*)p; p += SZ_HEADS;
    unsigned short* vb     = (unsigned short*)p; p += SZ_HEADS;
    unsigned short* yb     = (unsigned short*)p; p += SZ_HEADS;

    // 0) precision conversion + weight transposes (bf16)
    cvt_f32_bf16<<<(8192 * 768 + 255) / 256, 256, 0, stream>>>(x, xb, 8192 * 768);
    transpose_w<<<(768 * 2304 + 255) / 256, 256, 0, stream>>>(w_attn, wqkvT, 768, 2304);
    transpose_w<<<(768 * 768 + 255) / 256, 256, 0, stream>>>(w_proj, wprojT, 768, 768);

    // 1) QKV projection (Q pre-scaled by 1/sqrt(HS)):  blocks 256m x 128n
    qkv_gemm_kernel<<<dim3(18, 32), 256, 0, stream>>>(xb, wqkvT, b_attn, qb, kb, vb);

    // 2) causal flash attention: 96 heads x 32 query tiles / 8 waves per block
    flash_kernel<<<384, 256, 0, stream>>>(qb, kb, vb, yb);

    // 3) output projection -> f32:  blocks 256m x 128n
    proj_gemm_kernel<<<dim3(6, 32), 256, 0, stream>>>(yb, wprojT, b_proj, out);
}
